// RWKVAttention_53910429499507
// MI455X (gfx1250) — matrix operand order repaired and checked by reading, not compiled, not run on hardware
//
#include <hip/hip_runtime.h>
#include <hip/hip_bf16.h>
#include <math.h>

#define T_DIM 4096
#define HID   2048
#define NH    32
#define HD    64

typedef __attribute__((ext_vector_type(16))) __bf16 v16bf;
typedef __attribute__((ext_vector_type(8)))  __bf16 v8bf;
typedef __attribute__((ext_vector_type(8)))  float  v8f;

__device__ __forceinline__ __bf16 f32_to_bf16(float f) {
  union { float f; unsigned u; } a; a.f = f;
  unsigned r = a.u + 0x7FFFu + ((a.u >> 16) & 1u);   // round-to-nearest-even
  unsigned short h = (unsigned short)(r >> 16);
  union { unsigned short s; __bf16 b; } o; o.s = h;
  return o.b;
}

__device__ __forceinline__ float readlane_f(float v, int l) {
  return __int_as_float(__builtin_amdgcn_readlane(__float_as_int(v), l));
}

// ---------------------------------------------------------------------------
// f32 -> bf16 convert (weights)
// ---------------------------------------------------------------------------
__global__ void cvt_f32_bf16_kernel(const float* __restrict__ in,
                                    __bf16* __restrict__ out, int n) {
  int i = blockIdx.x * blockDim.x + threadIdx.x;
  if (i < n) out[i] = f32_to_bf16(in[i]);
}

// ---------------------------------------------------------------------------
// token shift + 4-way mix, emit bf16 activations for the WMMA GEMMs
// ---------------------------------------------------------------------------
__global__ void mix_kernel(const float* __restrict__ x,
                           const float* __restrict__ tmk,
                           const float* __restrict__ tmv,
                           const float* __restrict__ tmr,
                           const float* __restrict__ tmg,
                           __bf16* __restrict__ xk, __bf16* __restrict__ xv,
                           __bf16* __restrict__ xr, __bf16* __restrict__ xg) {
  int i = blockIdx.x * blockDim.x + threadIdx.x;   // over T*HID
  int t = i / HID;
  int c = i - t * HID;
  float xc   = x[i];
  float prev = (t > 0) ? x[i - HID] : 0.0f;
  float xx   = prev - xc;
  xk[i] = f32_to_bf16(xc + xx * (1.0f - tmk[c]));
  xv[i] = f32_to_bf16(xc + xx * (1.0f - tmv[c]));
  xr[i] = f32_to_bf16(xc + xx * (1.0f - tmr[c]));
  xg[i] = f32_to_bf16(xc + xx * (1.0f - tmg[c]));
}

// ---------------------------------------------------------------------------
// NT GEMM: C[M,N] = A[M,K] * B[N,K]^T   (A,B bf16 row-major, C fp32)
// Wave computes 32(M)x64(N) via 2x4 v_wmma_f32_16x16x32_bf16 tiles
// (8 WMMAs per K-step for 6 fragment loads). Block = 8 waves (4M x 2N)
// -> 128x128 block tile. mode==1 applies silu to the output.
// ---------------------------------------------------------------------------
__global__ __launch_bounds__(256)
void gemm_bf16_nt_kernel(const __bf16* __restrict__ A,
                         const __bf16* __restrict__ B,
                         float* __restrict__ C,
                         int M, int N, int K, int mode) {
  const int lane  = threadIdx.x & 31;
  const int wave  = threadIdx.x >> 5;       // 0..7
  const int wm    = wave & 3;               // 4 waves along M
  const int wn    = wave >> 2;              // 2 waves along N
  const int m0    = blockIdx.y * 128 + wm * 32;
  const int n0    = blockIdx.x * 128 + wn * 64;
  const int mrow  = lane & 15;
  const int khalf = lane >> 4;              // 0 or 1

  v8f acc[2][4] = {};

  const __bf16* Ap[2];
  Ap[0] = A + (size_t)(m0 + mrow)      * K;
  Ap[1] = A + (size_t)(m0 + 16 + mrow) * K;
  const __bf16* Bp[4];
  #pragma unroll
  for (int j = 0; j < 4; ++j)
    Bp[j] = B + (size_t)(n0 + j * 16 + mrow) * K;

  for (int kk = 0; kk < K; kk += 32) {
    if (kk + 512 < K) {
      __builtin_prefetch((const void*)(Ap[0] + kk + 512), 0, 0);
      __builtin_prefetch((const void*)(Bp[0] + kk + 512), 0, 0);
      __builtin_prefetch((const void*)(Bp[2] + kk + 512), 0, 0);
    }
    // A fragment k order used here (contiguous, same as B): lane<16 -> K 0..15,
    // lane>=16 -> K 16..31, all for row M=lane&15.
    union { v16bf v; v8bf h[2]; } af[2], bf[4];
    #pragma unroll
    for (int i = 0; i < 2; ++i) {
      af[i].h[0] = *(const v8bf*)(Ap[i] + kk +     khalf * 16);
      af[i].h[1] = *(const v8bf*)(Ap[i] + kk + 8 + khalf * 16);
    }
    // B fragment k order used here (contiguous, same as A): lane<16 -> K 0..15,
    // lane>=16 -> K 16..31, column N=lane&15.
    #pragma unroll
    for (int j = 0; j < 4; ++j)
      bf[j].v = *(const v16bf*)(Bp[j] + kk + khalf * 16);

    #pragma unroll
    for (int i = 0; i < 2; ++i)
      #pragma unroll
      for (int j = 0; j < 4; ++j)
        acc[i][j] = __builtin_amdgcn_wmma_f32_16x16x32_bf16(
            false, af[i].v, false, bf[j].v, (short)0, acc[i][j], false, false);
  }

  // C/D layout: VGPR rr, lanes 0-15 -> M=rr, lanes 16-31 -> M=8+rr; N=lane&15.
  #pragma unroll
  for (int i = 0; i < 2; ++i)
    #pragma unroll
    for (int j = 0; j < 4; ++j)
      #pragma unroll
      for (int rr = 0; rr < 8; ++rr) {
        float val = acc[i][j][rr];
        if (mode == 1) val = val / (1.0f + expf(-val));   // silu
        int row = m0 + i * 16 + khalf * 8 + rr;
        int col = n0 + j * 16 + mrow;
        C[(size_t)row * N + col] = val;
      }
}

// ---------------------------------------------------------------------------
// Sequential WKV scan + fused GroupNorm + gate, ONE wave32 per head.
// Lane e owns state columns e and e+32 (128 fp32 regs). Broadcasts of
// r[d], k[d], w[d] via v_readlane (constant lane after unroll); w readlanes
// are t-invariant and hoist into SGPRs. No barriers, no LDS. Next-step
// r/k/v/g loads are double-buffered to hide memory latency behind the
// 64-deep FMA chain. Writes bf16 activation 'a' for the final Wo GEMM.
// ---------------------------------------------------------------------------
__global__ __launch_bounds__(32)
void wkv_gn_kernel(const float* __restrict__ r, const float* __restrict__ k,
                   const float* __restrict__ v, const float* __restrict__ g,
                   const float* __restrict__ td, const float* __restrict__ tf,
                   const float* __restrict__ gamma, const float* __restrict__ beta,
                   __bf16* __restrict__ a) {
  const int h = blockIdx.x;
  const int e = threadIdx.x;                 // 0..31
  const int c0 = h * HD + e, c1 = c0 + 32;

  const float wv0 = expf(-expf(td[c0])), wv1 = expf(-expf(td[c1]));
  const float uv0 = tf[c0], uv1 = tf[c1];
  const float gm0 = gamma[c0], gm1 = gamma[c1];
  const float bt0 = beta[c0],  bt1 = beta[c1];

  float S0[HD], S1[HD];
  #pragma unroll
  for (int d = 0; d < HD; ++d) { S0[d] = 0.0f; S1[d] = 0.0f; }

  size_t base = (size_t)h * HD;
  float rv0 = r[base + e], rv1 = r[base + e + 32];
  float kv0 = k[base + e], kv1 = k[base + e + 32];
  float ve0 = v[base + e], ve1 = v[base + e + 32];
  float gv0 = g[base + e], gv1 = g[base + e + 32];

  for (int t = 0; t < T_DIM; ++t) {
    const size_t nbase = base + HID;
    float nrv0 = 0.f, nrv1 = 0.f, nkv0 = 0.f, nkv1 = 0.f;
    float nve0 = 0.f, nve1 = 0.f, ngv0 = 0.f, ngv1 = 0.f;
    if (t + 1 < T_DIM) {          // prefetch next step's operands early
      nrv0 = r[nbase + e]; nrv1 = r[nbase + e + 32];
      nkv0 = k[nbase + e]; nkv1 = k[nbase + e + 32];
      nve0 = v[nbase + e]; nve1 = v[nbase + e + 32];
      ngv0 = g[nbase + e]; ngv1 = g[nbase + e + 32];
    }

    float acc0 = 0.0f, acc1 = 0.0f;
    #pragma unroll
    for (int d = 0; d < 32; ++d) {
      const float srd = readlane_f(rv0, d);
      const float skd = readlane_f(kv0, d);
      const float swd = readlane_f(wv0, d);  // loop-invariant -> SGPR
      acc0  = fmaf(srd, S0[d], acc0);        // cross uses pre-update S
      acc1  = fmaf(srd, S1[d], acc1);
      S0[d] = fmaf(S0[d], swd, skd * ve0);   // decay + rank-1 update
      S1[d] = fmaf(S1[d], swd, skd * ve1);
    }
    #pragma unroll
    for (int d = 0; d < 32; ++d) {
      const float srd = readlane_f(rv1, d);
      const float skd = readlane_f(kv1, d);
      const float swd = readlane_f(wv1, d);
      acc0       = fmaf(srd, S0[32 + d], acc0);
      acc1       = fmaf(srd, S1[32 + d], acc1);
      S0[32 + d] = fmaf(S0[32 + d], swd, skd * ve0);
      S1[32 + d] = fmaf(S1[32 + d], swd, skd * ve1);
    }

    // diagonal (bonus) term: sum_d u[d]*r[d]*k[d] via wave butterfly
    float ud = fmaf(uv0 * rv0, kv0, uv1 * rv1 * kv1);
    #pragma unroll
    for (int off = 16; off > 0; off >>= 1) ud += __shfl_xor(ud, off, 32);

    float out0 = fmaf(ud, ve0, acc0) * 0.125f;   // / HEAD_SIZE_DIVISOR
    float out1 = fmaf(ud, ve1, acc1) * 0.125f;

    // GroupNorm over this head's 64 channels (2 per lane x 32 lanes)
    float s1 = out0 + out1;
    float s2 = fmaf(out0, out0, out1 * out1);
    #pragma unroll
    for (int off = 16; off > 0; off >>= 1) {
      s1 += __shfl_xor(s1, off, 32);
      s2 += __shfl_xor(s2, off, 32);
    }
    float mu  = s1 * (1.0f / HD);
    float var = s2 * (1.0f / HD) - mu * mu;
    float inv = rsqrtf(var + 1e-5f);

    a[base + e]      = f32_to_bf16(((out0 - mu) * inv * gm0 + bt0) * gv0);
    a[base + e + 32] = f32_to_bf16(((out1 - mu) * inv * gm1 + bt1) * gv1);

    base = nbase;
    rv0 = nrv0; rv1 = nrv1; kv0 = nkv0; kv1 = nkv1;
    ve0 = nve0; ve1 = nve1; gv0 = ngv0; gv1 = ngv1;
  }
}

// ---------------------------------------------------------------------------
extern "C" void kernel_launch(void* const* d_in, const int* in_sizes, int n_in,
                              void* d_out, int out_size, void* d_ws, size_t ws_size,
                              hipStream_t stream) {
  (void)in_sizes; (void)n_in; (void)out_size; (void)ws_size;

  const float* x    = (const float*)d_in[0];
  const float* Wr   = (const float*)d_in[1];
  const float* Wk   = (const float*)d_in[2];
  const float* Wv   = (const float*)d_in[3];
  const float* Wg   = (const float*)d_in[4];
  const float* Wo   = (const float*)d_in[5];
  const float* tmk  = (const float*)d_in[6];
  const float* tmv  = (const float*)d_in[7];
  const float* tmr  = (const float*)d_in[8];
  const float* tmg  = (const float*)d_in[9];
  const float* td   = (const float*)d_in[10];
  const float* tf   = (const float*)d_in[11];
  const float* gam  = (const float*)d_in[12];
  const float* bet  = (const float*)d_in[13];

  const size_t W_SZ = (size_t)HID * HID;     // 4M elems
  const size_t X_SZ = (size_t)T_DIM * HID;   // 8M elems

  char* ws = (char*)d_ws;
  size_t off = 0;
  __bf16* Wr_b = (__bf16*)(ws + off); off += W_SZ * 2;
  __bf16* Wk_b = (__bf16*)(ws + off); off += W_SZ * 2;
  __bf16* Wv_b = (__bf16*)(ws + off); off += W_SZ * 2;
  __bf16* Wg_b = (__bf16*)(ws + off); off += W_SZ * 2;
  __bf16* Wo_b = (__bf16*)(ws + off); off += W_SZ * 2;
  __bf16* xk_b = (__bf16*)(ws + off); off += X_SZ * 2;
  __bf16* xv_b = (__bf16*)(ws + off); off += X_SZ * 2;
  __bf16* xr_b = (__bf16*)(ws + off); off += X_SZ * 2;
  __bf16* xg_b = (__bf16*)(ws + off); off += X_SZ * 2;
  __bf16* a_b  = (__bf16*)(ws + off); off += X_SZ * 2;
  float*  r_f  = (float*)(ws + off);  off += X_SZ * 4;
  float*  k_f  = (float*)(ws + off);  off += X_SZ * 4;
  float*  v_f  = (float*)(ws + off);  off += X_SZ * 4;
  float*  g_f  = (float*)(ws + off);  off += X_SZ * 4;

  // 1) weights -> bf16
  int wgrid = (int)(W_SZ / 256);
  cvt_f32_bf16_kernel<<<wgrid, 256, 0, stream>>>(Wr, Wr_b, (int)W_SZ);
  cvt_f32_bf16_kernel<<<wgrid, 256, 0, stream>>>(Wk, Wk_b, (int)W_SZ);
  cvt_f32_bf16_kernel<<<wgrid, 256, 0, stream>>>(Wv, Wv_b, (int)W_SZ);
  cvt_f32_bf16_kernel<<<wgrid, 256, 0, stream>>>(Wg, Wg_b, (int)W_SZ);
  cvt_f32_bf16_kernel<<<wgrid, 256, 0, stream>>>(Wo, Wo_b, (int)W_SZ);

  // 2) token-shift mix -> bf16 activations
  mix_kernel<<<(int)(X_SZ / 256), 256, 0, stream>>>(
      x, tmk, tmv, tmr, tmg, xk_b, xv_b, xr_b, xg_b);

  // 3) four projection GEMMs (bf16 WMMA, fp32 out); g gets fused silu
  dim3 gg(HID / 128, T_DIM / 128);
  gemm_bf16_nt_kernel<<<gg, 256, 0, stream>>>(xk_b, Wk_b, k_f, T_DIM, HID, HID, 0);
  gemm_bf16_nt_kernel<<<gg, 256, 0, stream>>>(xv_b, Wv_b, v_f, T_DIM, HID, HID, 0);
  gemm_bf16_nt_kernel<<<gg, 256, 0, stream>>>(xr_b, Wr_b, r_f, T_DIM, HID, HID, 0);
  gemm_bf16_nt_kernel<<<gg, 256, 0, stream>>>(xg_b, Wg_b, g_f, T_DIM, HID, HID, 1);

  // 4) sequential WKV scan + GroupNorm + gate -> bf16 'a' (1 wave per head)
  wkv_gn_kernel<<<NH, 32, 0, stream>>>(r_f, k_f, v_f, g_f, td, tf, gam, bet, a_b);

  // 5) output projection: out = a @ Wo^T (fp32 into d_out)
  gemm_bf16_nt_kernel<<<gg, 256, 0, stream>>>(a_b, Wo_b, (float*)d_out,
                                              T_DIM, HID, HID, 0);
}
